// GAT_graph_branch_46127948759399
// MI455X (gfx1250) — matrix operand order repaired and checked by reading, not compiled, not run on hardware
//
#include <hip/hip_runtime.h>

typedef __attribute__((ext_vector_type(16))) _Float16 v16h;
typedef __attribute__((ext_vector_type(8)))  _Float16 v8h;
typedef __attribute__((ext_vector_type(8)))  float    v8f;

// ---- order-preserving float <-> uint encoding (for atomicMax on floats) ----
__device__ __forceinline__ unsigned enc_f(float f) {
    unsigned u = __float_as_uint(f);
    return (u & 0x80000000u) ? ~u : (u | 0x80000000u);
}
__device__ __forceinline__ float dec_f(unsigned u) {
    return __uint_as_float((u & 0x80000000u) ? (u & 0x7FFFFFFFu) : ~u);
}
#define ENC_NEG_INF 0x007FFFFFu   // enc_f(-inf)

// ---- f32 -> f16 bulk convert (once per activation tensor) ----
__global__ void cvt_f16_kernel(const float* __restrict__ in, _Float16* __restrict__ out,
                               long long n)
{
    long long t = (long long)blockIdx.x * blockDim.x + threadIdx.x;
    if (t < n) out[t] = (_Float16)in[t];
}

// =====================================================================
//  WMMA GEMM:  C[n, M] = A[n, K] @ W[K, M] + bias[M]
//  A is pre-converted f16. W is pre-swizzled into fragment-major LDS so a
//  lane's 16-half B fragment is one contiguous 32B run (2x ds_load_b128).
//  A fragments (K/32 of them) are register-resident, reused over N tiles.
// =====================================================================
template<int K, int M>
__global__ __launch_bounds__(256)
void gemm_wmma(const _Float16* __restrict__ A, const float* __restrict__ W,
               const float* __restrict__ bias, float* __restrict__ C, int nrows)
{
    static_assert(K % 32 == 0 && M % 16 == 0, "tile shape");
    constexpr int KC = K / 32;       // k-chunks
    constexpr int NT = M / 16;       // n-tiles

    // fragment-major: Ws[((nt*KC + kc)*32 + lane)*16 + j]
    //   value = W[k0 + laneHi*16 + j][nt*16 + laneLo]
    __shared__ _Float16 Ws[K * M];
    for (int i = threadIdx.x; i < K * M; i += 256) {
        const int k = i / M, n = i % M;
        const int kc = k >> 5, kin = k & 31;
        const int lane = ((kin >> 4) << 4) | (n & 15);
        const int j = kin & 15;
        const int nt = n >> 4;
        Ws[(((nt * KC + kc) * 32) + lane) * 16 + j] = (_Float16)W[i];
    }
    __syncthreads();

    const int lane   = threadIdx.x & 31;
    const int wave   = threadIdx.x >> 5;
    const int laneLo = lane & 15;
    const int laneHi = lane >> 4;
    const int row0   = (blockIdx.x * 8 + wave) * 16;
    if (row0 >= nrows) return;

    const _Float16* Arow = A + (size_t)(row0 + laneLo) * K;

    // A fragments: contiguous k = laneHi*16 + j (matches Ws); two aligned 16B loads each
    v16h a[KC];
#pragma unroll
    for (int kc = 0; kc < KC; ++kc) {
        const _Float16* p = Arow + kc * 32 + laneHi * 16;
        const v8h lo = *(const v8h*)(p);
        const v8h hi = *(const v8h*)(p + 8);
        a[kc] = __builtin_shufflevector(lo, hi, 0, 1, 2, 3, 4, 5, 6, 7,
                                        8, 9, 10, 11, 12, 13, 14, 15);
    }

#pragma unroll
    for (int nt = 0; nt < NT; ++nt) {
        v8f acc = {};
#pragma unroll
        for (int kc = 0; kc < KC; ++kc) {
            const v16h b = *(const v16h*)&Ws[(((nt * KC + kc) * 32) + lane) * 16];
            acc = __builtin_amdgcn_wmma_f32_16x16x32_f16(false, a[kc], false, b,
                                                         (short)0, acc, false, false);
        }
        const float bv = bias ? bias[nt * 16 + laneLo] : 0.f;
#pragma unroll
        for (int r = 0; r < 8; ++r) {
            const int orow = row0 + laneHi * 8 + r;   // D layout: VGPR r -> M=r (+8 hi half)
            C[(size_t)orow * M + nt * 16 + laneLo] = acc[r] + bv;
        }
    }
}

// =====================================================================
//  edge_attr mean (self-loop fill_value='mean') and e_loop vectors
// =====================================================================
__global__ void ea_sum_kernel(const float* __restrict__ ea, float* __restrict__ easum,
                              long long n_elems)
{
    long long t = (long long)blockIdx.x * blockDim.x + threadIdx.x;
    if (t >= n_elems) return;
    atomicAdd(&easum[t & 15], ea[t]);
}

__global__ void eloop_kernel(const float* __restrict__ easum,
                             const float* __restrict__ We1, const float* __restrict__ We2,
                             float* __restrict__ eloop1, float* __restrict__ eloop2,
                             int n_edges)
{
    int j = threadIdx.x;
    if (j < 64) {
        float a = 0.f;
        for (int k = 0; k < 16; ++k) a += (easum[k] / (float)n_edges) * We1[k * 64 + j];
        eloop1[j] = a;
    } else if (j < 96) {
        int j2 = j - 64;
        float a = 0.f;
        for (int k = 0; k < 16; ++k) a += (easum[k] / (float)n_edges) * We2[k * 32 + j2];
        eloop2[j2] = a;
    }
}

__global__ void fill_u32_kernel(unsigned* __restrict__ p, unsigned v, int n)
{
    int t = blockIdx.x * blockDim.x + threadIdx.x;
    if (t < n) p[t] = v;
}

// =====================================================================
//  GATv2 attention, phase 1: logits + segment max (encoded atomicMax)
// =====================================================================
template<int H, int C>
__global__ __launch_bounds__(256)
void attn_logits_kernel(const float* __restrict__ xl, const float* __restrict__ xr,
                        const float* __restrict__ edge_attr,
                        const float* __restrict__ We, const float* __restrict__ eloop,
                        const float* __restrict__ att,
                        const int* __restrict__ src, const int* __restrict__ dst,
                        float* __restrict__ alpha, unsigned* __restrict__ amax_enc,
                        int n_edges, int n_nodes)
{
    __shared__ float We_s[16 * H * C];
    __shared__ float att_s[H * C];
    __shared__ float el_s[H * C];
    for (int i = threadIdx.x; i < 16 * H * C; i += blockDim.x) We_s[i] = We[i];
    for (int i = threadIdx.x; i < H * C; i += blockDim.x) { att_s[i] = att[i]; el_s[i] = eloop[i]; }
    __syncthreads();

    const int e = blockIdx.x * blockDim.x + threadIdx.x;
    const int tot = n_edges + n_nodes;
    if (e >= tot) return;

    const bool selfloop = (e >= n_edges);
    int s, d;
    float ea[16] = {};
    if (selfloop) {
        s = d = e - n_edges;
    } else {
        s = src[e]; d = dst[e];
#pragma unroll
        for (int k = 0; k < 16; ++k) ea[k] = edge_attr[(size_t)e * 16 + k];
    }
    const float* xls = xl + (size_t)s * (H * C);
    const float* xrd = xr + (size_t)d * (H * C);

#pragma unroll
    for (int h = 0; h < H; ++h) {
        float acc = 0.f;
        for (int c = 0; c < C; ++c) {
            const int j = h * C + c;
            float ev;
            if (selfloop) {
                ev = el_s[j];
            } else {
                ev = 0.f;
#pragma unroll
                for (int k = 0; k < 16; ++k) ev += ea[k] * We_s[k * (H * C) + j];
            }
            float v = xls[j] + xrd[j] + ev;
            v = v > 0.f ? v : 0.2f * v;          // leaky_relu(0.2)
            acc += v * att_s[j];
        }
        alpha[(size_t)e * H + h] = acc;
        atomicMax(&amax_enc[d * H + h], enc_f(acc));
    }
}

// phase 2: exp(alpha - max) and segment-sum of denominators
template<int H>
__global__ void attn_exp_kernel(float* __restrict__ alpha, const int* __restrict__ dst,
                                const unsigned* __restrict__ amax_enc,
                                float* __restrict__ denom, int n_edges, int n_nodes)
{
    long long t = (long long)blockIdx.x * blockDim.x + threadIdx.x;
    long long tot = (long long)(n_edges + n_nodes) * H;
    if (t >= tot) return;
    const int e = (int)(t / H), h = (int)(t % H);
    const int d = (e >= n_edges) ? (e - n_edges) : dst[e];
    const float m = dec_f(amax_enc[d * H + h]);
    const float ex = __expf(alpha[t] - m);
    alpha[t] = ex;
    atomicAdd(&denom[d * H + h], ex);
}

// phase 3: out[dst] += xl[src] * alpha / denom   (one thread per edge-channel)
template<int H, int C>
__global__ void attn_agg_kernel(const float* __restrict__ xl, const float* __restrict__ alpha,
                                const float* __restrict__ denom,
                                const int* __restrict__ src, const int* __restrict__ dst,
                                float* __restrict__ out, int n_edges, int n_nodes)
{
    long long t = (long long)blockIdx.x * blockDim.x + threadIdx.x;
    long long tot = (long long)(n_edges + n_nodes) * (H * C);
    if (t >= tot) return;
    const int j = (int)(t % (H * C));
    const int e = (int)(t / (H * C));
    const int h = j / C;
    int s, d;
    if (e >= n_edges) { s = d = e - n_edges; } else { s = src[e]; d = dst[e]; }
    const float coeff = alpha[(size_t)e * H + h] / (denom[d * H + h] + 1e-16f);
    atomicAdd(&out[(size_t)d * (H * C) + j], xl[(size_t)s * (H * C) + j] * coeff);
}

// =====================================================================
//  BatchNorm (training stats) + skip add + ELU; OUT selects f32 or f16
// =====================================================================
template<int D>
__global__ void bn_stats_kernel(const float* __restrict__ h, float* __restrict__ st, int n)
{
    long long t = (long long)blockIdx.x * blockDim.x + threadIdx.x;
    if (t >= (long long)n * D) return;
    const int c = (int)(t % D);
    const float v = h[t];
    atomicAdd(&st[c], v);
    atomicAdd(&st[D + c], v * v);
}

template<int D, typename OUT>
__global__ void bn_apply_kernel(const float* __restrict__ h, const float* __restrict__ st,
                                const float* __restrict__ g, const float* __restrict__ b,
                                const float* __restrict__ hproj, OUT* __restrict__ out, int n)
{
    long long t = (long long)blockIdx.x * blockDim.x + threadIdx.x;
    if (t >= (long long)n * D) return;
    const int c = (int)(t % D);
    const float mean = st[c] / (float)n;
    const float var  = st[D + c] / (float)n - mean * mean;
    float v = (h[t] - mean) * rsqrtf(var + 1e-5f) * g[c] + b[c] + hproj[t];
    v = v > 0.f ? v : (__expf(v) - 1.f);         // ELU
    out[t] = (OUT)v;
}

// =====================================================================
//  Global mean+max pooling over graphs
// =====================================================================
__global__ void pool_count_kernel(const int* __restrict__ batch, float* __restrict__ counts, int n)
{
    int t = blockIdx.x * blockDim.x + threadIdx.x;
    if (t < n) atomicAdd(&counts[batch[t]], 1.f);
}

__global__ void pool_summax_kernel(const float* __restrict__ h2, const int* __restrict__ batch,
                                   float* __restrict__ psum, unsigned* __restrict__ pmax, int n)
{
    long long t = (long long)blockIdx.x * blockDim.x + threadIdx.x;
    if (t >= (long long)n * 32) return;
    const int c = (int)(t % 32), i = (int)(t / 32);
    const int gidx = batch[i];
    const float v = h2[t];
    atomicAdd(&psum[gidx * 32 + c], v);
    atomicMax(&pmax[gidx * 32 + c], enc_f(v));
}

__global__ void pool_final_kernel(const float* __restrict__ psum, const unsigned* __restrict__ pmax,
                                  const float* __restrict__ counts, float* __restrict__ out)
{
    int t = threadIdx.x;                       // 1024 = 16 graphs * 64 cols
    const int gidx = t / 64, c = t % 64;
    const float cnt = counts[gidx];
    float v;
    if (c < 32) v = psum[gidx * 32 + c] / fmaxf(cnt, 1.f);
    else        v = (cnt > 0.f) ? dec_f(pmax[gidx * 32 + (c - 32)]) : 0.f;
    out[t] = v;
}

// =====================================================================
//  Host orchestration
// =====================================================================
extern "C" void kernel_launch(void* const* d_in, const int* in_sizes, int n_in,
                              void* d_out, int out_size, void* d_ws, size_t ws_size,
                              hipStream_t stream)
{
    (void)n_in; (void)out_size; (void)ws_size;
    const float* x        = (const float*)d_in[0];
    const int*   eindex   = (const int*)  d_in[1];
    const float* ea       = (const float*)d_in[2];
    const int*   batch    = (const int*)  d_in[3];
    const float* skip1_W  = (const float*)d_in[4];
    const float* skip1_b  = (const float*)d_in[5];
    const float* c1_Wl    = (const float*)d_in[6];
    const float* c1_bl    = (const float*)d_in[7];
    const float* c1_Wr    = (const float*)d_in[8];
    const float* c1_br    = (const float*)d_in[9];
    const float* c1_We    = (const float*)d_in[10];
    const float* c1_att   = (const float*)d_in[11];
    /* conv1_bias (d_in[12]) cancels through BatchNorm -> unused */
    const float* bn1_g    = (const float*)d_in[13];
    const float* bn1_b    = (const float*)d_in[14];
    const float* skip2_W  = (const float*)d_in[15];
    const float* skip2_b  = (const float*)d_in[16];
    const float* c2_Wl    = (const float*)d_in[17];
    const float* c2_bl    = (const float*)d_in[18];
    const float* c2_Wr    = (const float*)d_in[19];
    const float* c2_br    = (const float*)d_in[20];
    const float* c2_We    = (const float*)d_in[21];
    const float* c2_att   = (const float*)d_in[22];
    /* conv2_bias (d_in[23]) cancels through BatchNorm -> unused */
    const float* bn2_g    = (const float*)d_in[24];
    const float* bn2_b    = (const float*)d_in[25];

    const int NN = in_sizes[0] / 128;          // 50000 nodes
    const int NE = in_sizes[1] / 2;            // 800000 edges
    const int NT = NE + NN;                    // with self-loops
    const int* src = eindex;
    const int* dst = eindex + NE;

    // ---- workspace layout (float units; f16 buffers counted as n/2) ----
    float* base = (float*)d_ws;
    size_t off = 0;
    auto alloc = [&](size_t nfloats) { float* p = base + off; off += nfloats; return p; };
    float*     xl1    = alloc((size_t)NN * 64);
    float*     xr1    = alloc((size_t)NN * 64);
    float*     hp1    = alloc((size_t)NN * 64);
    float*     h1     = alloc((size_t)NN * 64);
    float*     xl2    = alloc((size_t)NN * 32);
    float*     xr2    = alloc((size_t)NN * 32);
    float*     hp2    = alloc((size_t)NN * 32);
    float*     h2     = alloc((size_t)NN * 32);
    _Float16*  x_h    = (_Float16*)alloc((size_t)NN * 64);   // NN*128 halves
    _Float16*  h1_h   = (_Float16*)alloc((size_t)NN * 32);   // NN*64 halves
    float*     alpha1 = alloc((size_t)NT * 2);
    float*     alpha2 = alloc((size_t)NT);
    float*     denom1 = alloc((size_t)NN * 2);
    float*     denom2 = alloc((size_t)NN);
    unsigned*  amax1  = (unsigned*)alloc((size_t)NN * 2);
    unsigned*  amax2  = (unsigned*)alloc((size_t)NN);
    float*     easum  = alloc(16);
    float*     eloop1 = alloc(64);
    float*     eloop2 = alloc(32);
    float*     bns1   = alloc(128);
    float*     bns2   = alloc(64);
    float*     counts = alloc(16);
    float*     psum   = alloc(512);
    unsigned*  pmax   = (unsigned*)alloc(512);

    auto cdiv = [](long long a, long long b) { return (int)((a + b - 1) / b); };

    // ---- zero / init accumulators (graph-capture legal) ----
    hipMemsetAsync(h1,     0, (size_t)NN * 64 * 4, stream);
    hipMemsetAsync(h2,     0, (size_t)NN * 32 * 4, stream);
    hipMemsetAsync(denom1, 0, (size_t)NN * 2 * 4, stream);
    hipMemsetAsync(denom2, 0, (size_t)NN * 4, stream);
    hipMemsetAsync(easum,  0, 16 * 4, stream);
    hipMemsetAsync(bns1,   0, 128 * 4, stream);
    hipMemsetAsync(bns2,   0, 64 * 4, stream);
    hipMemsetAsync(counts, 0, 16 * 4, stream);
    hipMemsetAsync(psum,   0, 512 * 4, stream);
    fill_u32_kernel<<<cdiv(NN * 2, 256), 256, 0, stream>>>(amax1, ENC_NEG_INF, NN * 2);
    fill_u32_kernel<<<cdiv(NN, 256),     256, 0, stream>>>(amax2, ENC_NEG_INF, NN);
    fill_u32_kernel<<<1, 256, 0, stream>>>(pmax, ENC_NEG_INF, 512);

    // ---- self-loop edge-attr features (fill_value='mean') ----
    ea_sum_kernel<<<cdiv((long long)NE * 16, 256), 256, 0, stream>>>(ea, easum, (long long)NE * 16);
    eloop_kernel<<<1, 128, 0, stream>>>(easum, c1_We, c2_We, eloop1, eloop2, NE);

    // ---- x -> f16 once; then layer-1 WMMA GEMMs (K=128, M=64) ----
    cvt_f16_kernel<<<cdiv((long long)NN * 128, 256), 256, 0, stream>>>(x, x_h, (long long)NN * 128);
    const int g1 = cdiv(cdiv(NN, 16), 8);
    gemm_wmma<128, 64><<<g1, 256, 0, stream>>>(x_h, skip1_W, skip1_b, hp1, NN);
    gemm_wmma<128, 64><<<g1, 256, 0, stream>>>(x_h, c1_Wl,   c1_bl,   xl1, NN);
    gemm_wmma<128, 64><<<g1, 256, 0, stream>>>(x_h, c1_Wr,   c1_br,   xr1, NN);

    // ---- layer 1 attention (H=2, C=32) ----
    attn_logits_kernel<2, 32><<<cdiv(NT, 256), 256, 0, stream>>>(
        xl1, xr1, ea, c1_We, eloop1, c1_att, src, dst, alpha1, amax1, NE, NN);
    attn_exp_kernel<2><<<cdiv((long long)NT * 2, 256), 256, 0, stream>>>(
        alpha1, dst, amax1, denom1, NE, NN);
    attn_agg_kernel<2, 32><<<cdiv((long long)NT * 64, 256), 256, 0, stream>>>(
        xl1, alpha1, denom1, src, dst, h1, NE, NN);

    // ---- BN1 + skip + ELU; output written directly as f16 for layer 2 ----
    bn_stats_kernel<64><<<cdiv((long long)NN * 64, 256), 256, 0, stream>>>(h1, bns1, NN);
    bn_apply_kernel<64, _Float16><<<cdiv((long long)NN * 64, 256), 256, 0, stream>>>(
        h1, bns1, bn1_g, bn1_b, hp1, h1_h, NN);

    // ---- layer 2 WMMA GEMMs (K=64, M=32) ----
    gemm_wmma<64, 32><<<g1, 256, 0, stream>>>(h1_h, skip2_W, skip2_b, hp2, NN);
    gemm_wmma<64, 32><<<g1, 256, 0, stream>>>(h1_h, c2_Wl,   c2_bl,   xl2, NN);
    gemm_wmma<64, 32><<<g1, 256, 0, stream>>>(h1_h, c2_Wr,   c2_br,   xr2, NN);

    // ---- layer 2 attention (H=1, C=32) ----
    attn_logits_kernel<1, 32><<<cdiv(NT, 256), 256, 0, stream>>>(
        xl2, xr2, ea, c2_We, eloop2, c2_att, src, dst, alpha2, amax2, NE, NN);
    attn_exp_kernel<1><<<cdiv((long long)NT, 256), 256, 0, stream>>>(
        alpha2, dst, amax2, denom2, NE, NN);
    attn_agg_kernel<1, 32><<<cdiv((long long)NT * 32, 256), 256, 0, stream>>>(
        xl2, alpha2, denom2, src, dst, h2, NE, NN);

    // ---- BN2 + skip + ELU (f32, in place) ----
    bn_stats_kernel<32><<<cdiv((long long)NN * 32, 256), 256, 0, stream>>>(h2, bns2, NN);
    bn_apply_kernel<32, float><<<cdiv((long long)NN * 32, 256), 256, 0, stream>>>(
        h2, bns2, bn2_g, bn2_b, hp2, h2, NN);

    // ---- pooling ----
    pool_count_kernel<<<cdiv(NN, 256), 256, 0, stream>>>(batch, counts, NN);
    pool_summax_kernel<<<cdiv((long long)NN * 32, 256), 256, 0, stream>>>(h2, batch, psum, pmax, NN);
    pool_final_kernel<<<1, 1024, 0, stream>>>(psum, pmax, counts, (float*)d_out);
}
